// CrossAttentionModule_83958020702535
// MI455X (gfx1250) — matrix operand order repaired and checked by reading, not compiled, not run on hardware
//
#include <hip/hip_runtime.h>
#include <hip/hip_fp16.h>

typedef __attribute__((ext_vector_type(16))) __bf16 v16bf;
typedef __attribute__((ext_vector_type(8)))  float  v8f;
typedef __attribute__((ext_vector_type(4)))  float  f4;

#define BATCH 512
#define CCH   640
#define TT    196
#define HH    32
#define TPAD  208      // 13*16 padded token count
#define KTIL  13
#define NTIL  169      // 13*13 output tiles
#define NKC   20       // 640 / 32
#define LDA   40       // staged bf16 row stride (bank stride 20)
#define CLD   210      // corr f16 row stride (odd dword stride -> conflict-free columns)
#define NWAVE 16
#define NTPW  11       // ceil(169/16) tiles per wave
#define TINV  40.0f    // 1 / 0.025

__global__ __launch_bounds__(512)
void fused_xattn(const float* __restrict__ cf, const float* __restrict__ qf,
                 const float* __restrict__ cw1, const float* __restrict__ cb1,
                 const float* __restrict__ cw2, const float* __restrict__ cb2,
                 const float* __restrict__ qw1, const float* __restrict__ qb1,
                 const float* __restrict__ qw2, const float* __restrict__ qb2,
                 float* __restrict__ out)
{
  __shared__ __half                           s_corr[TT*CLD];
  __shared__ __attribute__((aligned(16))) __bf16 s_a[TPAD*LDA];
  __shared__ __attribute__((aligned(16))) __bf16 s_b[TPAD*LDA];
  __shared__ float s_invc[TT], s_invq[TT];
  __shared__ float s_mc[TT],  s_mq[TT];
  __shared__ float s_h[2*HH];
  __shared__ float s_kc[TT],  s_kq[TT];
  __shared__ float s_lc[TT],  s_lq[TT];
  __shared__ float s_ac[TT],  s_aq[TT];
  __shared__ float s_red[4];

  const int tid  = threadIdx.x;
  const int lane = tid & 31;
  const int wave = tid >> 5;
  const int half = lane >> 4;   // lane 16-31 -> upper half per ISA layout
  const int lrow = lane & 15;
  const int b    = blockIdx.x;
  const size_t base = (size_t)b * CCH * TT;

  // ---------- Phase 0: inverse L2 norms over channels (class & query) ----------
  for (int idx = tid; idx < 2*TT; idx += blockDim.x) {
    const int t = (idx < TT) ? idx : idx - TT;
    const float* src = (idx < TT) ? (cf + base) : (qf + base);
    float s = 0.f;
    for (int c = 0; c < CCH; ++c) { float v = src[c*TT + t]; s += v*v; }
    const float inv = 1.f / fmaxf(sqrtf(s), 1e-12f);
    if (idx < TT) s_invc[t] = inv; else s_invq[t] = inv;
  }

  // ---------- Phase 1: corr = A·Bᵀ (raw tokens) via bf16 WMMA ----------
  v8f acc[NTPW];
  #pragma unroll
  for (int s = 0; s < NTPW; ++s) acc[s] = (v8f){0,0,0,0,0,0,0,0};

  for (int kc = 0; kc < NKC; ++kc) {
    __syncthreads();
    // cooperative staging: f32 [c][t] global -> bf16 [t][k] LDS (pad rows zeroed)
    for (int i = tid; i < 2*32*TPAD; i += blockDim.x) {
      const int which = (i >= 32*TPAD);
      const int j  = which ? (i - 32*TPAD) : i;
      const int kk = j / TPAD;
      const int t  = j - kk*TPAD;
      const float* src = which ? (qf + base) : (cf + base);
      const float v = (t < TT) ? src[(kc*32 + kk)*TT + t] : 0.f;
      (which ? s_b : s_a)[t*LDA + kk] = (__bf16)v;
    }
    __syncthreads();

    #pragma unroll
    for (int s = 0; s < NTPW; ++s) {
      const int tile = wave + s*NWAVE;      // wave-uniform guard: EXEC stays all-1s
      if (tile < NTIL) {
        const int ti = tile / KTIL;
        const int tj = tile - ti*KTIL;
        union { v16bf v; f4 q[2]; } fa, fb;
        // A fragment: 16-bit A 16x32 layout: k runs [8h..8h+7] and [16+8h..23+8h]
        const __bf16* ap = &s_a[(ti*16 + lrow)*LDA];
        fa.q[0] = *(const f4*)(ap + half*8);
        fa.q[1] = *(const f4*)(ap + 16 + half*8);
        // B fragment: 16-bit B 32x16 layout: k runs [8h..8h+7] and [16+8h..23+8h]
        const __bf16* bp = &s_b[(tj*16 + lrow)*LDA];
        fb.q[0] = *(const f4*)(bp + half*8);
        fb.q[1] = *(const f4*)(bp + 16 + half*8);
        acc[s] = __builtin_amdgcn_wmma_f32_16x16x32_bf16(
                    false, fa.v, false, fb.v, (short)0, acc[s], false, false);
      }
    }
  }
  __syncthreads();

  // ---------- Phase 1.5: apply separable normalization, corr -> LDS (f16) ----------
  #pragma unroll
  for (int s = 0; s < NTPW; ++s) {
    const int tile = wave + s*NWAVE;
    if (tile < NTIL) {
      const int ti = tile / KTIL;
      const int tj = tile - ti*KTIL;
      const int u  = tj*16 + lrow;
      const float iq = (u < TT) ? s_invq[u] : 0.f;
      #pragma unroll
      for (int r = 0; r < 8; ++r) {           // C/D layout: VGPR r -> M = r + 8*half
        const int t = ti*16 + r + half*8;
        if (t < TT && u < TT)
          s_corr[t*CLD + u] = (__half)(acc[s][r] * s_invc[t] * iq);
      }
    }
  }
  __syncthreads();

  // ---------- Phase 2a: means (class: column mean; query: row mean) ----------
  for (int idx = tid; idx < 2*TT; idx += blockDim.x) {
    if (idx < TT) {
      const int u = idx; float s = 0.f;
      for (int t = 0; t < TT; ++t) s += (float)s_corr[t*CLD + u];
      s_mc[u] = s * (1.f/TT);
    } else {
      const int t = idx - TT; float s = 0.f;
      for (int u = 0; u < TT; ++u) s += (float)s_corr[t*CLD + u];
      s_mq[t] = s * (1.f/TT);
    }
  }
  __syncthreads();

  // ---------- Phase 2b: MLP hidden (relu(m @ w1 + b1)), 64 threads ----------
  if (tid < 2*HH) {
    const int which = tid >= HH;
    const int j = which ? tid - HH : tid;
    const float* w1 = which ? qw1 : cw1;
    const float* b1 = which ? qb1 : cb1;
    const float* m  = which ? s_mq : s_mc;
    float s = b1[j];
    for (int t = 0; t < TT; ++t) s += m[t] * w1[t*HH + j];
    s_h[tid] = fmaxf(s, 0.f);
  }
  __syncthreads();

  // ---------- Phase 2c: kernel vectors (h @ w2 + b2) ----------
  for (int idx = tid; idx < 2*TT; idx += blockDim.x) {
    const int which = idx >= TT;
    const int u = which ? idx - TT : idx;
    const float* w2 = which ? qw2 : cw2;
    const float* b2 = which ? qb2 : cb2;
    const float* h  = which ? (s_h + HH) : s_h;
    float s = b2[u];
    for (int j = 0; j < HH; ++j) s += h[j] * w2[j*TT + u];
    (which ? s_kq : s_kc)[u] = s;
  }
  __syncthreads();

  // ---------- Phase 2d: logits (class: row dot; query: column dot) ----------
  for (int idx = tid; idx < 2*TT; idx += blockDim.x) {
    if (idx < TT) {
      const int t = idx; float s = 0.f;
      for (int u = 0; u < TT; ++u) s += (float)s_corr[t*CLD + u] * s_kc[u];
      s_lc[t] = s * TINV;
    } else {
      const int u = idx - TT; float s = 0.f;
      for (int t = 0; t < TT; ++t) s += (float)s_corr[t*CLD + u] * s_kq[t];
      s_lq[u] = s * TINV;
    }
  }
  __syncthreads();

  // ---------- Phase 2e: softmax over tokens ----------
  if (tid < 2) {
    const float* l = tid ? s_lq : s_lc;
    float mx = -3.0e38f;
    for (int t = 0; t < TT; ++t) mx = fmaxf(mx, l[t]);
    float sm = 0.f;
    for (int t = 0; t < TT; ++t) sm += __expf(l[t] - mx);
    s_red[tid*2] = mx; s_red[tid*2+1] = 1.f / sm;
  }
  __syncthreads();
  for (int idx = tid; idx < 2*TT; idx += blockDim.x) {
    const int which = idx >= TT;
    const int t = which ? idx - TT : idx;
    const float* l = which ? s_lq : s_lc;
    (which ? s_aq : s_ac)[t] = __expf(l[t] - s_red[which*2]) * s_red[which*2+1];
  }
  __syncthreads();

  // ---------- Phase 3: out = feat * (1 + attn), float4 + non-temporal stores ----------
  const int NV = CCH * (TT/4);               // 31360 float4 per feature map
  const f4* csrc = (const f4*)(cf + base);
  const f4* qsrc = (const f4*)(qf + base);
  f4* cdst = (f4*)out + (size_t)b * NV;
  f4* qdst = (f4*)out + (size_t)BATCH * NV + (size_t)b * NV;
  for (int i = tid; i < 2*NV; i += blockDim.x) {
    const int which = i >= NV;
    const int j = which ? i - NV : i;
    const f4 v = which ? qsrc[j] : csrc[j];
    const float* at = which ? s_aq : s_ac;
    const int t = (j % 49) * 4;              // T=196 -> 49 float4 per (b,c) row
    f4 g;
    g.x = v.x * (1.f + at[t+0]);
    g.y = v.y * (1.f + at[t+1]);
    g.z = v.z * (1.f + at[t+2]);
    g.w = v.w * (1.f + at[t+3]);
    __builtin_nontemporal_store(g, (which ? qdst : cdst) + j);
  }
}

extern "C" void kernel_launch(void* const* d_in, const int* in_sizes, int n_in,
                              void* d_out, int out_size, void* d_ws, size_t ws_size,
                              hipStream_t stream) {
  const float* cf  = (const float*)d_in[0];
  const float* qf  = (const float*)d_in[1];
  const float* cw1 = (const float*)d_in[2];
  const float* cb1 = (const float*)d_in[3];
  const float* cw2 = (const float*)d_in[4];
  const float* cb2 = (const float*)d_in[5];
  const float* qw1 = (const float*)d_in[6];
  const float* qb1 = (const float*)d_in[7];
  const float* qw2 = (const float*)d_in[8];
  const float* qb2 = (const float*)d_in[9];
  float* out = (float*)d_out;
  fused_xattn<<<BATCH, 512, 0, stream>>>(cf, qf, cw1, cb1, cw2, cb2,
                                         qw1, qb1, qw2, qb2, out);
}
